// GraphToGraph_24824910971395
// MI455X (gfx1250) — matrix operand order repaired and checked by reading, not compiled, not run on hardware
//
#include <hip/hip_runtime.h>

typedef __attribute__((ext_vector_type(16))) _Float16 v16h;
typedef __attribute__((ext_vector_type(8)))  float    v8f;

#define CON_DIM   128
#define STRUC_DIM 64
#define HIDDEN    128
#define FDIM      192          // CON_DIM + STRUC_DIM
#define NUM_UPDATES 5
#define LP_ITERS    100

// ---------------------------------------------------------------- atomics
__device__ __forceinline__ void atomAddF(float* p, float v) {
  unsafeAtomicAdd(p, v);       // lowers to global_atomic_add_f32 on AMD
}

// ---------------------------------------------------------------- W1 pre-pack
// Pack W1 [256,128] into V_WMMA_F32_16X16X32_F16 B-fragment layout.
// Fragment id f = nt*8 + ks (nt = 16-wide N tile, ks = 32-wide K step).
// Lane l (<16): K = 32*ks + 0..15, N = nt*16 + l
// Lane l (>=16): K = 32*ks + 16..31, N = nt*16 + (l-16)
// v16h element 2j / 2j+1 = K offsets 2j / 2j+1 within the lane's 16-K range.
__global__ void pack_w1_kernel(const float* __restrict__ W1,
                               _Float16* __restrict__ pB) {
  int idx = blockIdx.x * blockDim.x + threadIdx.x;   // 64 frags * 32 lanes * 8 dwords
  if (idx >= 64 * 32 * 8) return;
  int j  = idx & 7;
  int l  = (idx >> 3) & 31;
  int f  = idx >> 8;
  int nt = f >> 3;
  int ks = f & 7;
  int k0 = 32 * ks + ((l < 16) ? 0 : 16) + 2 * j;
  int n  = nt * 16 + (l & 15);
  pB[idx * 2 + 0] = (_Float16)W1[(size_t)k0 * HIDDEN + n];
  pB[idx * 2 + 1] = (_Float16)W1[(size_t)(k0 + 1) * HIDDEN + n];
}

// ---------------------------------------------------------------- edge MLP (WMMA)
// One wave = 16 edges. A (16x32 f16) gathered from con[src]|con[dst] per the
// contiguous K order of pB: lane<16 holds M=lane, K = 32ks+0..15;
// lane>=16 holds M=lane-16, K = 32ks+16..31.
__global__ __launch_bounds__(256) void edge_mlp_kernel(
    const float* __restrict__ con,
    const _Float16* __restrict__ pB,
    const float* __restrict__ b1,
    const float* __restrict__ W2,
    const float* __restrict__ b2,
    const int* __restrict__ src,
    const int* __restrict__ dst,
    float* __restrict__ scores,
    int E) {
  const int lane = threadIdx.x & 31;
  const int wave = threadIdx.x >> 5;
  const int e0   = blockIdx.x * 128 + wave * 16;
  const int lm   = lane & 15;
  const int hi   = lane >> 4;

  int em  = e0 + lm;
  int emc = (em < E) ? em : 0;
  const int srow = src[emc];
  const int drow = dst[emc];
  const float* srcp = con + (size_t)srow * CON_DIM;
  const float* dstp = con + (size_t)drow * CON_DIM;

  // Gather all A fragments (K = 0..255) once.
  v16h Af[8];
#pragma unroll
  for (int ks = 0; ks < 8; ++ks) {
    int base1 = 32 * ks + (hi ? 16 : 0);
    int base2 = base1 + 8;
    const float* r1 = (base1 < CON_DIM) ? (srcp + base1) : (dstp + (base1 - CON_DIM));
    const float* r2 = (base2 < CON_DIM) ? (srcp + base2) : (dstp + (base2 - CON_DIM));
    float4 x0 = *(const float4*)(r1);
    float4 x1 = *(const float4*)(r1 + 4);
    float4 x2 = *(const float4*)(r2);
    float4 x3 = *(const float4*)(r2 + 4);
    v16h a;
    a[0]=(_Float16)x0.x;  a[1]=(_Float16)x0.y;  a[2]=(_Float16)x0.z;  a[3]=(_Float16)x0.w;
    a[4]=(_Float16)x1.x;  a[5]=(_Float16)x1.y;  a[6]=(_Float16)x1.z;  a[7]=(_Float16)x1.w;
    a[8]=(_Float16)x2.x;  a[9]=(_Float16)x2.y;  a[10]=(_Float16)x2.z; a[11]=(_Float16)x2.w;
    a[12]=(_Float16)x3.x; a[13]=(_Float16)x3.y; a[14]=(_Float16)x3.z; a[15]=(_Float16)x3.w;
    Af[ks] = a;
  }

  float p[8];
#pragma unroll
  for (int r = 0; r < 8; ++r) p[r] = 0.f;

  const v16h* pBv = (const v16h*)pB;

#pragma unroll
  for (int nt = 0; nt < 8; ++nt) {
    float bias = b1[nt * 16 + lm];     // all C elems in a lane share N
    v8f acc;
#pragma unroll
    for (int r = 0; r < 8; ++r) acc[r] = bias;
#pragma unroll
    for (int ks = 0; ks < 8; ++ks) {
      v16h B = pBv[(size_t)((nt * 8 + ks) * 32 + lane)];
      acc = __builtin_amdgcn_wmma_f32_16x16x32_f16(
          false, Af[ks], false, B, (short)0, acc, false, false);
    }
    // layer-2: relu(h) * W2[n], accumulate per row
    float w2v = W2[nt * 16 + lm];
#pragma unroll
    for (int r = 0; r < 8; ++r) {
      float h = acc[r] > 0.f ? acc[r] : 0.f;
      p[r] += h * w2v;
    }
  }

  // Reduce over the 16 lanes of each half (masks stay inside each half).
#pragma unroll
  for (int m = 1; m < 16; m <<= 1) {
#pragma unroll
    for (int r = 0; r < 8; ++r) p[r] += __shfl_xor(p[r], m, 32);
  }

  // lanes 0..7 -> rows 0..7 ; lanes 16..23 -> rows 8..15
  if (lm < 8) {
    int row = lm + (hi ? 8 : 0);
    int ew  = e0 + row;
    if (ew < E) {
      float raw = p[lm] + b2[0];
      float s   = raw > 0.f ? raw : 0.f;
      int ss = src[ew], dd = dst[ew];
      scores[ew] = (dd > ss) ? s : 0.f;
    }
  }
}

// ---------------------------------------------------------------- utility kernels
__global__ void fill_f32_kernel(float* p, float v, int n) {
  int i = blockIdx.x * blockDim.x + threadIdx.x;
  if (i < n) p[i] = v;
}
__global__ void fill_i32_kernel(int* p, int v, int n) {
  int i = blockIdx.x * blockDim.x + threadIdx.x;
  if (i < n) p[i] = v;
}
__global__ void iota_kernel(int* p, int n) {
  int i = blockIdx.x * blockDim.x + threadIdx.x;
  if (i < n) p[i] = i;
}
__global__ void build_feats_kernel(const float* __restrict__ con,
                                   const float* __restrict__ st,
                                   float* __restrict__ f, int n) {
  int idx = blockIdx.x * blockDim.x + threadIdx.x;
  if (idx >= n * FDIM) return;
  int i = idx / FDIM, c = idx - i * FDIM;
  f[idx] = (c < CON_DIM) ? con[(size_t)i * CON_DIM + c]
                         : st[(size_t)i * STRUC_DIM + (c - CON_DIM)];
}

// node_sums[i] = 1 (self loop) + sum of incident pruned-edge scores
__global__ void node_sum_kernel(const int* __restrict__ src, const int* __restrict__ dst,
                                const float* __restrict__ sc, float* ns, int E) {
  int e = blockIdx.x * blockDim.x + threadIdx.x;
  if (e >= E) return;
  float s = sc[e];
  if (s <= 0.f) return;
  atomAddF(&ns[src[e]], s);
  atomAddF(&ns[dst[e]], s);
}

// out[i] = f[i] / ns[i]   (the normalized self-loop term)
__global__ void spmm_init_kernel(const float* __restrict__ fin, const float* __restrict__ ns,
                                 float* __restrict__ fout, int total) {
  int idx = blockIdx.x * blockDim.x + threadIdx.x;
  if (idx >= total) return;
  fout[idx] = fin[idx] / ns[idx / FDIM];
}

// scatter both directions of each pruned edge; thread = (edge, 4-col chunk)
__global__ void spmm_edges_kernel(const int* __restrict__ src, const int* __restrict__ dst,
                                  const float* __restrict__ sc, const float* __restrict__ ns,
                                  const float* __restrict__ fin, float* __restrict__ fout,
                                  int E) {
  int t = blockIdx.x * blockDim.x + threadIdx.x;
  if (t >= E * 48) return;
  int e = t / 48, ch = t - e * 48;
  float s = sc[e];
  if (s <= 0.f) return;
  int a = src[e], b = dst[e];
  float wa = s / ns[a];
  float wb = s / ns[b];
  int c0 = ch * 4;
  float4 fb = *(const float4*)(fin + (size_t)b * FDIM + c0);
  float4 fa = *(const float4*)(fin + (size_t)a * FDIM + c0);
  float* oa = fout + (size_t)a * FDIM + c0;
  float* ob = fout + (size_t)b * FDIM + c0;
  atomAddF(oa + 0, wa * fb.x); atomAddF(oa + 1, wa * fb.y);
  atomAddF(oa + 2, wa * fb.z); atomAddF(oa + 3, wa * fb.w);
  atomAddF(ob + 0, wb * fa.x); atomAddF(ob + 1, wb * fa.y);
  atomAddF(ob + 2, wb * fa.z); atomAddF(ob + 3, wb * fa.w);
}

// ---- label propagation (synchronous, matches jax while_loop semantics) ----
__global__ void lp_copy_kernel(const int* gin, int* gout, int n) {
  int i = blockIdx.x * blockDim.x + threadIdx.x;
  if (i < n) gout[i] = gin[i];
}
__global__ void lp_edges_kernel(const int* __restrict__ src, const int* __restrict__ dst,
                                const float* __restrict__ sc,
                                const int* __restrict__ gin, int* __restrict__ gout, int E) {
  int e = blockIdx.x * blockDim.x + threadIdx.x;
  if (e >= E) return;
  if (sc[e] > 0.f) atomicMin(&gout[dst[e]], gin[src[e]]);
}

// ---- unique(return_inverse) via flag + exclusive scan ----
__global__ void mark_flags_kernel(const int* g, int* flags, int n) {
  int i = blockIdx.x * blockDim.x + threadIdx.x;
  if (i < n) flags[g[i]] = 1;
}
__global__ void scan_excl_kernel(const int* __restrict__ flags, int* __restrict__ rank, int n) {
  __shared__ int sh[1024];
  __shared__ int carry;
  if (threadIdx.x == 0) carry = 0;
  __syncthreads();
  for (int base = 0; base < n; base += 1024) {
    int i = base + (int)threadIdx.x;
    int v = (i < n) ? flags[i] : 0;
    sh[threadIdx.x] = v;
    __syncthreads();
    for (int off = 1; off < 1024; off <<= 1) {
      int t = (threadIdx.x >= (unsigned)off) ? sh[threadIdx.x - off] : 0;
      __syncthreads();
      sh[threadIdx.x] += t;
      __syncthreads();
    }
    int c = carry;
    if (i < n) rank[i] = c + sh[threadIdx.x] - v;   // exclusive
    __syncthreads();
    if (threadIdx.x == 1023) carry = c + sh[1023];
    __syncthreads();
  }
}
__global__ void make_inv_kernel(const int* g, const int* rank, int* inv, int n) {
  int i = blockIdx.x * blockDim.x + threadIdx.x;
  if (i < n) inv[i] = rank[g[i]];
}

// ---- group mean ----
__global__ void group_accum_kernel(const int* __restrict__ inv, const float* __restrict__ f,
                                   float* __restrict__ out, int* __restrict__ counts, int n) {
  int t = blockIdx.x * blockDim.x + threadIdx.x;
  if (t >= n * 48) return;
  int i = t / 48, ch = t - i * 48;
  int g = inv[i];
  if (ch == 0) atomicAdd(&counts[g], 1);
  int c0 = ch * 4;
  float4 v = *(const float4*)(f + (size_t)i * FDIM + c0);
  float* o = out + (size_t)g * FDIM + c0;
  atomAddF(o + 0, v.x); atomAddF(o + 1, v.y);
  atomAddF(o + 2, v.z); atomAddF(o + 3, v.w);
}
__global__ void group_div_kernel(const int* __restrict__ counts, float* out, int total) {
  int idx = blockIdx.x * blockDim.x + threadIdx.x;
  if (idx >= total) return;
  float d = (float)counts[idx / FDIM];
  out[idx] /= (d > 1.f ? d : 1.f);
}

// ---------------------------------------------------------------- launch
extern "C" void kernel_launch(void* const* d_in, const int* in_sizes, int n_in,
                              void* d_out, int out_size, void* d_ws, size_t ws_size,
                              hipStream_t stream) {
  const float* con = (const float*)d_in[0];
  const float* st  = (const float*)d_in[1];
  const float* W1  = (const float*)d_in[2];
  const float* b1  = (const float*)d_in[3];
  const float* W2  = (const float*)d_in[4];
  const float* b2  = (const float*)d_in[5];
  const int* eids  = (const int*)d_in[6];

  const int N = in_sizes[0] / CON_DIM;
  const int E = in_sizes[6] / 2;
  const int* src = eids;
  const int* dst = eids + E;
  float* out = (float*)d_out;

  char* wb = (char*)d_ws;
  auto carve = [&](size_t bytes) -> void* {
    void* p = (void*)wb;
    wb += (bytes + 255) & ~(size_t)255;
    return p;
  };
  float* scores   = (float*)carve((size_t)E * 4);
  float* ns       = (float*)carve((size_t)N * 4);
  float* f0       = (float*)carve((size_t)N * FDIM * 4);
  float* f1       = (float*)carve((size_t)N * FDIM * 4);
  int*   g0       = (int*)carve((size_t)N * 4);
  int*   g1       = (int*)carve((size_t)N * 4);
  int*   flags    = (int*)carve((size_t)N * 4);
  int*   rankbuf  = (int*)carve((size_t)N * 4);
  int*   invbuf   = (int*)carve((size_t)N * 4);
  int*   counts   = (int*)carve((size_t)N * 4);
  _Float16* pB    = (_Float16*)carve((size_t)64 * 32 * 16 * 2);

  const int T = 256;
  auto blocks = [&](long long n) { return (int)((n + T - 1) / T); };

  // 1. pack W1 into WMMA B-fragment layout (64 KB, L2-resident afterwards)
  pack_w1_kernel<<<blocks(64 * 32 * 8), T, 0, stream>>>(W1, pB);

  // 2. edge-scoring MLP via v_wmma_f32_16x16x32_f16
  edge_mlp_kernel<<<(E + 127) / 128, 256, 0, stream>>>(con, pB, b1, W2, b2,
                                                       src, dst, scores, E);

  // 3. row sums (self loop contributes 1.0)
  fill_f32_kernel<<<blocks(N), T, 0, stream>>>(ns, 1.0f, N);
  node_sum_kernel<<<blocks(E), T, 0, stream>>>(src, dst, scores, ns, E);

  // 4. combined features + 5 SPMM rounds (ping-pong)
  build_feats_kernel<<<blocks((long long)N * FDIM), T, 0, stream>>>(con, st, f0, N);
  float* fin = f0; float* fout = f1;
  for (int it = 0; it < NUM_UPDATES; ++it) {
    spmm_init_kernel<<<blocks((long long)N * FDIM), T, 0, stream>>>(fin, ns, fout, N * FDIM);
    spmm_edges_kernel<<<blocks((long long)E * 48), T, 0, stream>>>(src, dst, scores, ns,
                                                                   fin, fout, E);
    float* tmp = fin; fin = fout; fout = tmp;
  }

  // 5. synchronous min-label propagation (100 iters == reference while_loop)
  iota_kernel<<<blocks(N), T, 0, stream>>>(g0, N);
  int* gin = g0; int* gout = g1;
  for (int it = 0; it < LP_ITERS; ++it) {
    lp_copy_kernel<<<blocks(N), T, 0, stream>>>(gin, gout, N);
    lp_edges_kernel<<<blocks(E), T, 0, stream>>>(src, dst, scores, gin, gout, E);
    int* tmp = gin; gin = gout; gout = tmp;
  }

  // 6. compact labels (unique + return_inverse)
  fill_i32_kernel<<<blocks(N), T, 0, stream>>>(flags, 0, N);
  mark_flags_kernel<<<blocks(N), T, 0, stream>>>(gin, flags, N);
  scan_excl_kernel<<<1, 1024, 0, stream>>>(flags, rankbuf, N);
  make_inv_kernel<<<blocks(N), T, 0, stream>>>(gin, rankbuf, invbuf, N);

  // 7. per-group means -> d_out [N, 192] (rows beyond #groups stay zero)
  fill_i32_kernel<<<blocks(N), T, 0, stream>>>(counts, 0, N);
  fill_f32_kernel<<<blocks((long long)N * FDIM), T, 0, stream>>>(out, 0.f, N * FDIM);
  group_accum_kernel<<<blocks((long long)N * 48), T, 0, stream>>>(invbuf, fin, out, counts, N);
  group_div_kernel<<<blocks((long long)N * FDIM), T, 0, stream>>>(counts, out, N * FDIM);
}
